// YOLO_finn_20847771255252
// MI455X (gfx1250) — compile-verified, matrix operand order checked by reading, not run on hardware
//
#include <hip/hip_runtime.h>
#include <stdint.h>
#include <math.h>

typedef int v8i __attribute__((ext_vector_type(8)));

// ---------------- scratch layout (bytes) ----------------
static constexpr size_t OFF_PW1 = 0;                      // 64 x 576
static constexpr size_t OFF_PW2 = 36864;                  // 128 x 576
static constexpr size_t OFF_PW3 = 110592;                 // 128 x 1152
static constexpr size_t OFF_PW4 = 258048;                 // 256 x 1152
static constexpr size_t OFF_PW5 = 552960;                 // 32 x 2304
static constexpr size_t OFF_PW0 = 626688;                 // 64 x 27 int8 (padded)
static constexpr size_t OFF_XQ  = 628736;                 // int8 [8,3,416,416]
static constexpr size_t OFF_A0P = 4782080;                // int8 [8,208,208,64]
static constexpr size_t OFF_A1  = OFF_A0P + 22151168;     // int8 [8,208,208,64]
static constexpr size_t OFF_A1P = OFF_A1  + 22151168;     // int8 [8,104,104,64]
static constexpr size_t OFF_A2  = OFF_A1P + 5537792;      // int8 [8,104,104,128]
static constexpr size_t OFF_A2P = OFF_A2  + 11075584;     // int8 [8,52,52,128]
static constexpr size_t OFF_A3  = OFF_A2P + 2768896;      // int8 [8,52,52,128]
static constexpr size_t OFF_A4  = OFF_A3  + 2768896;      // int8 [8,52,52,256]
static constexpr size_t OFF_A5  = OFF_A4  + 5537792;      // float [8,52,52,32]

__constant__ float c_anch[10] = {10.f,13.f,16.f,30.f,33.f,23.f,62.f,45.f,116.f,90.f};

// ---------------- weight packing ----------------
// OIHW float -> int8 [oc][t*Cin+ic], 2-bit quant: clip(round(w),-1,1)
__global__ void pack_w2(const float* __restrict__ w, int8_t* __restrict__ pw,
                        int Cout, int Cin) {
  int idx = blockIdx.x * blockDim.x + threadIdx.x;
  int total = Cout * Cin * 9;
  if (idx >= total) return;
  int t  = idx % 9;
  int ic = (idx / 9) % Cin;
  int oc = idx / (9 * Cin);
  float v = w[(size_t)(oc * Cin + ic) * 9 + t];
  float q = fminf(fmaxf(rintf(v), -1.f), 1.f);
  pw[(size_t)oc * (9 * Cin) + t * Cin + ic] = (int8_t)(int)q;
}

// 8-bit quant with oc padding: clip(round(w*127),-127,127); rows >= CoutReal zero
__global__ void pack_w8(const float* __restrict__ w, int8_t* __restrict__ pw,
                        int CoutReal, int CoutPad, int Cin) {
  int idx = blockIdx.x * blockDim.x + threadIdx.x;
  int total = CoutPad * Cin * 9;
  if (idx >= total) return;
  int t  = idx % 9;
  int ic = (idx / 9) % Cin;
  int oc = idx / (9 * Cin);
  int8_t val = 0;
  if (oc < CoutReal) {
    float v = w[(size_t)(oc * Cin + ic) * 9 + t];
    val = (int8_t)(int)fminf(fmaxf(rintf(v * 127.f), -127.f), 127.f);
  }
  pw[(size_t)oc * (9 * Cin) + t * Cin + ic] = val;
}

// layer-0 weights (OIHW [64,3,3,3]) -> int8, same flat order
__global__ void pack_w0(const float* __restrict__ w, int8_t* __restrict__ pw) {
  int idx = blockIdx.x * blockDim.x + threadIdx.x;
  if (idx >= 64 * 27) return;
  pw[idx] = (int8_t)(int)fminf(fmaxf(rintf(w[idx]), -1.f), 1.f);
}

// input quant: int8 = clip(round((2x-1)*128), -128, 127), NCHW plane layout
__global__ void quant_in(const float* __restrict__ x, int8_t* __restrict__ xq) {
  int idx = blockIdx.x * blockDim.x + threadIdx.x;
  const int total = 8 * 3 * 416 * 416;
  if (idx >= total) return;
  float v = x[idx];
  xq[idx] = (int8_t)(int)fminf(fmaxf(rintf((2.f * v - 1.f) * 128.f), -128.f), 127.f);
}

// ---------------- layer 0: int conv + BN + qact2 + 2x2 maxpool ----------------
// xq: int8 NCHW [8,3,416,416]; out: int8 NHWC [8,208,208,64]
__global__ void conv0_fused(const int8_t* __restrict__ xq, const int8_t* __restrict__ pw0,
                            const float* __restrict__ g0, const float* __restrict__ b0,
                            int8_t* __restrict__ out) {
  int idx = blockIdx.x * blockDim.x + threadIdx.x;
  const int total = 8 * 208 * 208 * 64;
  if (idx >= total) return;
  int oc = idx & 63;
  int xo = (idx >> 6) % 208;
  int yo = ((idx >> 6) / 208) % 208;
  int n  = (idx >> 6) / (208 * 208);

  int wq[27];
  #pragma unroll
  for (int i = 0; i < 27; ++i) wq[i] = (int)pw0[oc * 27 + i];

  const float rsq = rsqrtf(1.0f + 1e-4f);
  float ga = g0[oc] * rsq * (1.f / 128.f), bb = b0[oc];

  int best = -1000;
  #pragma unroll
  for (int py = 0; py < 2; ++py)
    #pragma unroll
    for (int px = 0; px < 2; ++px) {
      int y = 2 * yo + py, xc = 2 * xo + px;
      int s = 0;
      #pragma unroll
      for (int ic = 0; ic < 3; ++ic) {
        const int8_t* xp = xq + ((size_t)(n * 3 + ic)) * 416 * 416;
        #pragma unroll
        for (int ky = 0; ky < 3; ++ky) {
          int yy = y + ky - 1;
          if (yy < 0 || yy >= 416) continue;
          #pragma unroll
          for (int kx = 0; kx < 3; ++kx) {
            int xx = xc + kx - 1;
            if (xx < 0 || xx >= 416) continue;
            s += (int)xp[(size_t)yy * 416 + xx] * wq[ic * 9 + ky * 3 + kx];
          }
        }
      }
      float f = (float)s * ga + bb;
      int q = (int)fminf(fmaxf(rintf(f * 2.f), -2.f), 1.f);
      best = best > q ? best : q;
    }
  out[idx] = (int8_t)best;
}

// ---------------- int8 NHWC 2x2 maxpool ----------------
__global__ void maxpool2(const int8_t* __restrict__ in, int8_t* __restrict__ out,
                         int N, int H, int W, int C) {
  int Ho = H >> 1, Wo = W >> 1;
  long long total = (long long)N * Ho * Wo * C;
  long long idx = (long long)blockIdx.x * blockDim.x + threadIdx.x;
  if (idx >= total) return;
  int c  = (int)(idx % C);
  int xo = (int)((idx / C) % Wo);
  int yo = (int)((idx / C / Wo) % Ho);
  int n  = (int)(idx / C / Wo / Ho);
  const int8_t* p = in + (((size_t)n * H + 2 * yo) * W + 2 * xo) * C + c;
  int a = p[0], b = p[C];
  int d = p[(size_t)W * C], e = p[(size_t)W * C + C];
  int m = a > b ? a : b;
  m = m > d ? m : d;
  m = m > e ? m : e;
  out[idx] = (int8_t)m;
}

// ---------------- implicit-GEMM 3x3 conv: WMMA iu8 + LDS-staged weights ----------------
// in: int8 NHWC [N,H,W,CIN]; wp: int8 [CoutPad][9*CIN] with K = t*CIN+ic
// Block (8 waves): covers XW x-tiles * (8/XW) y-rows of one oc-block (16*NT channels).
// Per tap: stage (16*NT) x CIN weight slice into LDS (row pad 16B -> conflict-free
// ds_load_b64 B-fragment reads), then per 64-wide K-chunk do 1 A load + NT WMMAs.
template<int NT, bool QOUT, int CIN>
__global__ __launch_bounds__(256)
void conv3x3_wmma(const int8_t* __restrict__ in, const int8_t* __restrict__ wp,
                  const float* __restrict__ g, const float* __restrict__ bns,
                  void* __restrict__ outv,
                  int N, int H, int W, int Cout, float scale, int XW) {
  constexpr int OCB  = 16 * NT;
  constexpr int Ktot = 9 * CIN;
  constexpr int CinP = CIN + 16;              // padded LDS row stride
  constexpr int CPR  = CIN >> 4;              // 16B chunks per LDS row
  __shared__ int8_t smem[OCB * CinP];         // <= 9216 B

  const int XT = (W + 15) >> 4;
  const int XB = (XT + XW - 1) / XW;
  const int YW = 8 / XW;
  const int OB = Cout / OCB;

  int bid = blockIdx.x;
  int ob = bid % OB;
  int t1 = bid / OB;
  int xb = t1 % XB;
  int t2 = t1 / XB;
  int yb = t2 % (H / YW);
  int n  = t2 / (H / YW);
  int oc0 = ob * OCB;

  int wv = (int)threadIdx.x >> 5;
  int y  = yb * YW + wv / XW;
  int tx = xb * XW + wv % XW;
  int x0 = tx * 16;                            // may be >= W for dead waves (compute zeros)

  int lane = (int)threadIdx.x & 31;
  int r = lane & 15;       // A row (output pixel) / B column (oc) index
  int h = lane >> 4;       // K-half selector
  int bi = 2 * h;          // dword base for 8-bit fragment chunks

  v8i acc[NT];
  #pragma unroll
  for (int i = 0; i < NT; ++i)
    #pragma unroll
    for (int v = 0; v < 8; ++v) acc[i][v] = 0;

  int xa = x0 + r;

  for (int t = 0; t < 9; ++t) {
    // stage this tap's weight slice: OCB rows x CIN bytes
    __syncthreads();
    #pragma unroll
    for (int i = 0; i < (OCB * CPR + 255) / 256; ++i) {
      int idx = (int)threadIdx.x + i * 256;
      if (OCB * CPR % 256 == 0 || idx < OCB * CPR) {
        int row = idx / CPR, col = (idx % CPR) << 4;
        int4 v = *(const int4*)(wp + (size_t)(oc0 + row) * Ktot + t * CIN + col);
        *(int4*)(smem + row * CinP + col) = v;
      }
    }
    __syncthreads();

    int dy = t / 3 - 1, dx = t % 3 - 1;
    int yy = y + dy, xx = xa + dx;
    bool inb = (yy >= 0) && (yy < H) && (xx >= 0) && (xx < W);

    #pragma unroll
    for (int c0 = 0; c0 < CIN; c0 += 64) {
      v8i a;
      #pragma unroll
      for (int v = 0; v < 8; ++v) a[v] = 0;
      if (inb) {
        const int* p = (const int*)(in + (((size_t)n * H + yy) * W + xx) * CIN + c0);
        a[0] = p[bi];      a[1] = p[bi + 1];
        a[2] = p[bi + 4];  a[3] = p[bi + 5];
        a[4] = p[bi + 8];  a[5] = p[bi + 9];
        a[6] = p[bi + 12]; a[7] = p[bi + 13];
      }
      #pragma unroll
      for (int nb = 0; nb < NT; ++nb) {
        const int* q = (const int*)(smem + (nb * 16 + r) * CinP + c0);
        v8i bf;
        bf[0] = q[bi];      bf[1] = q[bi + 1];
        bf[2] = q[bi + 4];  bf[3] = q[bi + 5];
        bf[4] = q[bi + 8];  bf[5] = q[bi + 9];
        bf[6] = q[bi + 12]; bf[7] = q[bi + 13];
        acc[nb] = __builtin_amdgcn_wmma_i32_16x16x64_iu8(
            true, a, true, bf, acc[nb], false, false);
      }
    }
  }

  // Epilogue: D element (m = v + 8*h, ncol = lane&15)
  const float rsq = rsqrtf(1.0f + 1e-4f);
  #pragma unroll
  for (int nb = 0; nb < NT; ++nb) {
    int oc = oc0 + nb * 16 + (lane & 15);
    float ga = 0.f, bb = 0.f;
    if (QOUT) { ga = g[oc] * rsq; bb = bns[oc]; }
    #pragma unroll
    for (int v = 0; v < 8; ++v) {
      int m = v + 8 * h;
      int xp = x0 + m;
      if (xp < W) {
        float f = (float)acc[nb][v] * scale;
        size_t off = (((size_t)n * H + y) * W + xp) * Cout + oc;
        if (QOUT) {
          f = f * ga + bb;
          int qv = (int)fminf(fmaxf(rintf(f * 2.f), -2.f), 1.f);
          ((int8_t*)outv)[off] = (int8_t)qv;
        } else {
          ((float*)outv)[off] = f;
        }
      }
    }
  }
}

// ---------------- YOLO decode ----------------
// a5: float NHWC [8,52,52,32] (channels 0..29 valid, c = a*6+o)
// out: [io (8,5*52*52,6)] ++ [p (8,5,52,52,6)]
__global__ void yolo_decode(const float* __restrict__ a5, float* __restrict__ out) {
  int idx = blockIdx.x * blockDim.x + threadIdx.x;
  const int total = 8 * 5 * 52 * 52;
  if (idx >= total) return;
  int x = idx % 52;
  int y = (idx / 52) % 52;
  int a = (idx / (52 * 52)) % 5;
  int n = idx / (5 * 52 * 52);
  const float* p = a5 + (((size_t)n * 52 + y) * 52 + x) * 32 + a * 6;
  float p0 = p[0], p1 = p[1], p2 = p[2], p3 = p[3], p4 = p[4], p5 = p[5];
  float s0 = 1.f / (1.f + expf(-p0));
  float s1 = 1.f / (1.f + expf(-p1));
  float s4 = 1.f / (1.f + expf(-p4));
  float s5 = 1.f / (1.f + expf(-p5));
  size_t base = (((size_t)n * 5 + a) * 2704 + (size_t)y * 52 + x) * 6;
  out[base + 0] = (s0 + (float)x) * 8.f;
  out[base + 1] = (s1 + (float)y) * 8.f;
  out[base + 2] = expf(p2) * c_anch[2 * a];
  out[base + 3] = expf(p3) * c_anch[2 * a + 1];
  out[base + 4] = s4;
  out[base + 5] = s5;
  const size_t IOT = (size_t)8 * 5 * 2704 * 6;
  float* q = out + IOT + base;   // p flattens in the same (n,a,y,x,o) order
  q[0] = p0; q[1] = p1; q[2] = p2; q[3] = p3; q[4] = p4; q[5] = p5;
}

// ---------------- launch ----------------
static inline int cdiv(long long a, int b) { return (int)((a + b - 1) / b); }

extern "C" void kernel_launch(void* const* d_in, const int* in_sizes, int n_in,
                              void* d_out, int out_size, void* d_ws, size_t ws_size,
                              hipStream_t stream) {
  (void)in_sizes; (void)n_in; (void)out_size; (void)ws_size;
  const float* x  = (const float*)d_in[0];
  const float* w0 = (const float*)d_in[1];
  const float* w1 = (const float*)d_in[2];
  const float* w2 = (const float*)d_in[3];
  const float* w3 = (const float*)d_in[4];
  const float* w4 = (const float*)d_in[5];
  const float* w5 = (const float*)d_in[6];
  const float* g0 = (const float*)d_in[7];
  const float* g1 = (const float*)d_in[8];
  const float* g2 = (const float*)d_in[9];
  const float* g3 = (const float*)d_in[10];
  const float* g4 = (const float*)d_in[11];
  const float* b0 = (const float*)d_in[12];
  const float* b1 = (const float*)d_in[13];
  const float* b2 = (const float*)d_in[14];
  const float* b3 = (const float*)d_in[15];
  const float* b4 = (const float*)d_in[16];

  char* ws = (char*)d_ws;
  int8_t* pw1 = (int8_t*)(ws + OFF_PW1);
  int8_t* pw2 = (int8_t*)(ws + OFF_PW2);
  int8_t* pw3 = (int8_t*)(ws + OFF_PW3);
  int8_t* pw4 = (int8_t*)(ws + OFF_PW4);
  int8_t* pw5 = (int8_t*)(ws + OFF_PW5);
  int8_t* pw0 = (int8_t*)(ws + OFF_PW0);
  int8_t* xq  = (int8_t*)(ws + OFF_XQ);
  int8_t* a0p = (int8_t*)(ws + OFF_A0P);
  int8_t* a1  = (int8_t*)(ws + OFF_A1);
  int8_t* a1p = (int8_t*)(ws + OFF_A1P);
  int8_t* a2  = (int8_t*)(ws + OFF_A2);
  int8_t* a2p = (int8_t*)(ws + OFF_A2P);
  int8_t* a3  = (int8_t*)(ws + OFF_A3);
  int8_t* a4  = (int8_t*)(ws + OFF_A4);
  float*  a5  = (float*)(ws + OFF_A5);

  const int TB = 256;

  // packing / input quant
  pack_w2<<<cdiv(64LL * 64 * 9, TB), TB, 0, stream>>>(w1, pw1, 64, 64);
  pack_w2<<<cdiv(128LL * 64 * 9, TB), TB, 0, stream>>>(w2, pw2, 128, 64);
  pack_w2<<<cdiv(128LL * 128 * 9, TB), TB, 0, stream>>>(w3, pw3, 128, 128);
  pack_w2<<<cdiv(256LL * 128 * 9, TB), TB, 0, stream>>>(w4, pw4, 256, 128);
  pack_w8<<<cdiv(32LL * 256 * 9, TB), TB, 0, stream>>>(w5, pw5, 30, 32, 256);
  pack_w0<<<cdiv(64LL * 27, TB), TB, 0, stream>>>(w0, pw0);
  quant_in<<<cdiv(8LL * 3 * 416 * 416, TB), TB, 0, stream>>>(x, xq);

  // layer 0: int conv + BN + qact2 + pool -> a0p [8,208,208,64]
  conv0_fused<<<cdiv(8LL * 208 * 208 * 64, TB), TB, 0, stream>>>(xq, pw0, g0, b0, a0p);

  const float S24 = 0.5f;            // (a/2) * (w/1)
  const float S5  = 1.0f / 254.0f;   // (a/2) * (w/127)

  // layer 1: 208x208, 64->64   (XW=8 -> XB=2, YW=1)
  {
    int blocks = 8 * 208 * 2 * (64 / 64);
    conv3x3_wmma<4, true, 64><<<blocks, TB, 0, stream>>>(
        a0p, pw1, g1, b1, a1, 8, 208, 208, 64, S24, 8);
    maxpool2<<<cdiv(8LL * 104 * 104 * 64, TB), TB, 0, stream>>>(a1, a1p, 8, 208, 208, 64);
  }
  // layer 2: 104x104, 64->128  (XW=8 -> XB=1, YW=1)
  {
    int blocks = 8 * 104 * 1 * (128 / 64);
    conv3x3_wmma<4, true, 64><<<blocks, TB, 0, stream>>>(
        a1p, pw2, g2, b2, a2, 8, 104, 104, 128, S24, 8);
    maxpool2<<<cdiv(8LL * 52 * 52 * 128, TB), TB, 0, stream>>>(a2, a2p, 8, 104, 104, 128);
  }
  // layer 3: 52x52, 128->128   (XW=4 -> XB=1, YW=2, YB=26)
  {
    int blocks = 8 * 26 * 1 * (128 / 64);
    conv3x3_wmma<4, true, 128><<<blocks, TB, 0, stream>>>(
        a2p, pw3, g3, b3, a3, 8, 52, 52, 128, S24, 4);
  }
  // layer 4: 52x52, 128->256
  {
    int blocks = 8 * 26 * 1 * (256 / 64);
    conv3x3_wmma<4, true, 128><<<blocks, TB, 0, stream>>>(
        a3, pw4, g4, b4, a4, 8, 52, 52, 256, S24, 4);
  }
  // layer 5: 52x52, 256->32(pad of 30), float out, no BN/quant
  {
    int blocks = 8 * 26 * 1 * (32 / 32);
    conv3x3_wmma<2, false, 256><<<blocks, TB, 0, stream>>>(
        a4, pw5, nullptr, nullptr, a5, 8, 52, 52, 32, S5, 4);
  }
  // YOLO decode -> d_out (io ++ p)
  yolo_decode<<<cdiv(8LL * 5 * 52 * 52, TB), TB, 0, stream>>>(a5, (float*)d_out);
}
